// GTrajRecovery_85298050499129
// MI455X (gfx1250) — compile-verified, matrix operand order checked by reading, not run on hardware
//
#include <hip/hip_runtime.h>
#include <hip/hip_bf16.h>
#include <stdint.h>

// ---------------------------------------------------------------------------
// GTrajRecovery for MI455X (gfx1250): all GEMMs via v_wmma_f32_16x16x32_f16.
// Weights converted+packed to f16 WMMA-fragment order once per launch; the
// sequential GRU scan runs 255 iterations of small WMMA GEMMs + VALU kernels.
// Each GEMM wave computes a 16x64 strip: one A-fragment feeds 4 WMMAs.
// ---------------------------------------------------------------------------

typedef __attribute__((ext_vector_type(16))) _Float16 v16h;
typedef __attribute__((ext_vector_type(8)))  float    v8f;

#define BB 256   // batch
#define LL 128   // route length
#define HH 256   // hidden
#define TT 256   // time steps
#define NNODE (BB*LL)

struct alignas(16) B16 { uint32_t w[4]; };   // 16-byte POD chunk for b128 loads

// K index covered by half `i` (0..15) of a 16-bit A/B fragment for lane half `hi`
__device__ __forceinline__ int frag_k(int i, int hi) {
  int r = i >> 1, p = i & 1;
  int k = (r < 4) ? (2*r + p) : (16 + 2*(r-4) + p);
  return k + 8*hi;
}

// ---------------- generic utility kernels ----------------
__global__ void zero_kernel(float* p, long n) {
  long i = (long)blockIdx.x * blockDim.x + threadIdx.x;
  if (i < n) p[i] = 0.f;
}

__global__ void cvt_f32_to_f16(const float* __restrict__ src, _Float16* __restrict__ dst, long n) {
  long i = (long)blockIdx.x * blockDim.x + threadIdx.x;
  if (i < n) dst[i] = (_Float16)src[i];
}

// Pack weight matrix Bmat[k][n] (k<Ksrc zero-padded to Kpad) into per-wave
// WMMA B-fragment order: dst[((nt*nKt + kt)*32 + lane)*16 + i]
__global__ void pack_weight_f16(const float* __restrict__ src, _Float16* __restrict__ dst,
                                int Ksrc, int Kpad, int N, int ld, int transpose) {
  int nKt = Kpad >> 5;
  long total = (long)(N >> 4) * nKt * 512;
  for (long idx = (long)blockIdx.x * blockDim.x + threadIdx.x; idx < total;
       idx += (long)gridDim.x * blockDim.x) {
    int i    = (int)(idx & 15);
    int lane = (int)((idx >> 4) & 31);
    int kt   = (int)((idx >> 9) % nKt);
    int nt   = (int)(idx / ((long)nKt << 9));
    int n = nt * 16 + (lane & 15);
    int k = kt * 32 + frag_k(i, lane >> 4);
    float v = 0.f;
    if (k < Ksrc) v = transpose ? src[(long)n * ld + k] : src[(long)k * ld + n];
    dst[idx] = (_Float16)v;
  }
}

// ---------------- WMMA GEMM: C[MxN] = A[MxKpad](f16,row-major) * Bpacked + bias.
// One wave computes a 16x64 strip: single A-fragment load feeds 4 WMMAs.
// Requires N % 64 == 0 (all call sites: 256 / 512 / 768).
__global__ void wmma_gemm4(const _Float16* __restrict__ A, int lda,
                           const _Float16* __restrict__ Bp,
                           const float* __restrict__ bias,
                           float* __restrict__ C, int ldc,
                           int M, int N, int Kpad, int relu) {
  int wave = blockIdx.x * (blockDim.x >> 5) + (threadIdx.x >> 5);
  int lane = threadIdx.x & 31;
  int ngrp = N >> 6;                      // groups of four 16-wide n-tiles
  int mt = wave / ngrp;
  int ng = wave - mt * ngrp;
  if (mt * 16 >= M) return;
  int nKt = Kpad >> 5;
  int hi = lane >> 4;
  const _Float16* arow  = A + (long)(mt * 16 + (lane & 15)) * lda + 8 * hi;
  const _Float16* bbase = Bp + (((long)ng * 4) * nKt << 9) + lane * 16;
  const long btile = (long)nKt << 9;      // halves per packed n-tile
  v8f acc0 = {}, acc1 = {}, acc2 = {}, acc3 = {};
  for (int kt = 0; kt < nKt; ++kt) {
    v16h a, b0, b1, b2, b3;
    reinterpret_cast<B16*>(&a)[0]  = *reinterpret_cast<const B16*>(arow + kt*32);
    reinterpret_cast<B16*>(&a)[1]  = *reinterpret_cast<const B16*>(arow + kt*32 + 16);
    const _Float16* bk = bbase + (long)kt * 512;
    reinterpret_cast<B16*>(&b0)[0] = *reinterpret_cast<const B16*>(bk);
    reinterpret_cast<B16*>(&b0)[1] = *reinterpret_cast<const B16*>(bk + 8);
    reinterpret_cast<B16*>(&b1)[0] = *reinterpret_cast<const B16*>(bk + btile);
    reinterpret_cast<B16*>(&b1)[1] = *reinterpret_cast<const B16*>(bk + btile + 8);
    reinterpret_cast<B16*>(&b2)[0] = *reinterpret_cast<const B16*>(bk + 2*btile);
    reinterpret_cast<B16*>(&b2)[1] = *reinterpret_cast<const B16*>(bk + 2*btile + 8);
    reinterpret_cast<B16*>(&b3)[0] = *reinterpret_cast<const B16*>(bk + 3*btile);
    reinterpret_cast<B16*>(&b3)[1] = *reinterpret_cast<const B16*>(bk + 3*btile + 8);
    acc0 = __builtin_amdgcn_wmma_f32_16x16x32_f16(false, a, false, b0, (short)0, acc0, false, false);
    acc1 = __builtin_amdgcn_wmma_f32_16x16x32_f16(false, a, false, b1, (short)0, acc1, false, false);
    acc2 = __builtin_amdgcn_wmma_f32_16x16x32_f16(false, a, false, b2, (short)0, acc2, false, false);
    acc3 = __builtin_amdgcn_wmma_f32_16x16x32_f16(false, a, false, b3, (short)0, acc3, false, false);
  }
  int row0 = mt * 16 + 8 * hi;
  int ncol0 = ng * 64 + (lane & 15);
#pragma unroll
  for (int j = 0; j < 4; ++j) {
    v8f acc = (j == 0) ? acc0 : (j == 1) ? acc1 : (j == 2) ? acc2 : acc3;
    int ncol = ncol0 + j * 16;
    float bv = bias ? bias[ncol] : 0.f;
#pragma unroll
    for (int v = 0; v < 8; ++v) {
      float val = acc[v] + bv;
      if (relu) val = fmaxf(val, 0.f);
      C[(long)(row0 + v) * ldc + ncol] = val;
    }
  }
}

// ---------------- GAT: attention coefficients a_s / a_d ----------------
__global__ void gat_attn_coef(const float* __restrict__ xw,
                              const float* __restrict__ a_src, const float* __restrict__ a_dst,
                              float* __restrict__ as_out, float* __restrict__ ad_out) {
  int idx = blockIdx.x * blockDim.x + threadIdx.x;   // node*4 + head
  if (idx >= NNODE * 4) return;
  int node = idx >> 2, hd = idx & 3;
  const float* row = xw + (long)node * HH + hd * 64;
  float ss = 0.f, sd = 0.f;
  for (int d = 0; d < 64; ++d) {
    float v = row[d];
    ss += v * a_src[hd * 64 + d];
    sd += v * a_dst[hd * 64 + d];
  }
  as_out[idx] = ss; ad_out[idx] = sd;
}

// ---------------- GAT combine (chain neighbors) + residual + LayerNorm ----
__global__ void gat_combine_ln(const float* __restrict__ xw,
                               const float* __restrict__ a_s, const float* __restrict__ a_d,
                               const float* __restrict__ x_in,   // route_emb
                               const float* __restrict__ bias,
                               const float* __restrict__ gamma, const float* __restrict__ beta,
                               float* __restrict__ out32, _Float16* __restrict__ out16) {
  int node = blockIdx.x;          // 0..NNODE-1
  int l = node & (LL - 1);
  int t = threadIdx.x;            // 0..255
  __shared__ float al0[4], al1[4];
  __shared__ float red[HH];
  __shared__ float mean_s, var_s;
  if (t < 4) {
    int hd = t;
    float ad = a_d[node * 4 + hd];
    bool has0 = l > 0, has1 = l < (LL - 1);
    float e0 = -1e30f, e1 = -1e30f;
    if (has0) { float e = a_s[(node - 1) * 4 + hd] + ad; e0 = e > 0.f ? e : 0.2f * e; }
    if (has1) { float e = a_s[(node + 1) * 4 + hd] + ad; e1 = e > 0.f ? e : 0.2f * e; }
    float mx = fmaxf(e0, e1);
    float x0 = has0 ? __expf(e0 - mx) : 0.f;
    float x1 = has1 ? __expf(e1 - mx) : 0.f;
    float den = x0 + x1 + 1e-16f;
    al0[hd] = x0 / den; al1[hd] = x1 / den;
  }
  __syncthreads();
  int hd = t >> 6;
  float v = bias[t];
  if (l > 0)        v += al0[hd] * xw[(long)(node - 1) * HH + t];
  if (l < (LL - 1)) v += al1[hd] * xw[(long)(node + 1) * HH + t];
  float pre = x_in[(long)node * HH + t] + v;
  red[t] = pre; __syncthreads();
  for (int s = HH / 2; s > 0; s >>= 1) { if (t < s) red[t] += red[t + s]; __syncthreads(); }
  if (t == 0) mean_s = red[0] * (1.f / HH);
  __syncthreads();
  float dv = pre - mean_s;
  red[t] = dv * dv; __syncthreads();
  for (int s = HH / 2; s > 0; s >>= 1) { if (t < s) red[t] += red[t + s]; __syncthreads(); }
  if (t == 0) var_s = red[0] * (1.f / HH);
  __syncthreads();
  float o = dv * rsqrtf(var_s + 1e-5f) * gamma[t] + beta[t];
  out32[(long)node * HH + t] = o;
  out16[(long)node * HH + t] = (_Float16)o;
}

// ---------------- hidden0 = mean over L ----------------
__global__ void hidden_init(const float* __restrict__ route32,
                            float* __restrict__ h, _Float16* __restrict__ h16) {
  int idx = blockIdx.x * blockDim.x + threadIdx.x;   // b*H + j
  if (idx >= BB * HH) return;
  int b = idx >> 8, j = idx & 255;
  float s = 0.f;
  for (int l = 0; l < LL; ++l) s += route32[((long)(b * LL + l)) * HH + j];
  s *= (1.f / LL);
  h[idx] = s; h16[idx] = (_Float16)s;
}

// ---------------- per-step: build x = [emb_id[tid] , rate, 0-pad] (f16, 288) ----
__global__ void build_x(const int* __restrict__ trg_id, const float* __restrict__ trg_rate,
                        const float* __restrict__ emb_id, _Float16* __restrict__ x16, int t) {
  int idx = blockIdx.x * blockDim.x + threadIdx.x;   // b*288 + c
  if (idx >= BB * 288) return;
  int b = idx / 288, c = idx - b * 288;
  float v = 0.f;
  if (c < HH) { int id = trg_id[t * BB + b]; v = emb_id[(long)id * HH + c]; }
  else if (c == HH) v = trg_rate[t * BB + b];
  x16[idx] = (_Float16)v;
}

// ---------------- per-step: GRU gate combine ----------------
__global__ void gru_combine(const float* __restrict__ gi, const float* __restrict__ gh,
                            float* __restrict__ h, _Float16* __restrict__ h16) {
  int idx = blockIdx.x * blockDim.x + threadIdx.x;   // b*H + j
  if (idx >= BB * HH) return;
  int b = idx >> 8, j = idx & 255;
  const float* gib = gi + (long)b * 768;
  const float* ghb = gh + (long)b * 768;
  float r = 1.f / (1.f + __expf(-(gib[j]       + ghb[j])));
  float z = 1.f / (1.f + __expf(-(gib[256 + j] + ghb[256 + j])));
  float n = tanhf(gib[512 + j] + r * ghb[512 + j]);
  float hp = h[idx];
  float hn = (1.f - z) * n + z * hp;
  h[idx] = hn; h16[idx] = (_Float16)hn;
}

// ---------------- per-step: attention + softmax + weighted sum ----------------
__global__ void attn_step(const float* __restrict__ h, const _Float16* __restrict__ route16,
                          const int* __restrict__ mask, float* __restrict__ out_id,
                          _Float16* __restrict__ rate_in, int t) {
  int b = blockIdx.x, tt = threadIdx.x;               // 256 threads
  __shared__ float sc[LL];
  __shared__ float red[LL];
  __shared__ float mx_s, sum_s;
  const float* hb = h + (long)b * HH;
  if (tt < LL) {
    const _Float16* rr = route16 + ((long)(b * LL + tt)) * HH;
    float s = 0.f;
    for (int k = 0; k < HH; ++k) s += hb[k] * (float)rr[k];
    s *= 0.0625f;                                     // 1/sqrt(256)
    if (mask[b * LL + tt] == 0) s = -1e9f;
    sc[tt] = s; red[tt] = s;
  }
  __syncthreads();
  for (int s = LL / 2; s > 0; s >>= 1) { if (tt < s) red[tt] = fmaxf(red[tt], red[tt + s]); __syncthreads(); }
  if (tt == 0) mx_s = red[0];
  __syncthreads();
  if (tt < LL) { float e = __expf(sc[tt] - mx_s); sc[tt] = e; red[tt] = e; }
  __syncthreads();
  for (int s = LL / 2; s > 0; s >>= 1) { if (tt < s) red[tt] += red[tt + s]; __syncthreads(); }
  if (tt == 0) sum_s = red[0];
  __syncthreads();
  float inv = 1.f / sum_s;
  if (tt < LL) {
    float a = sc[tt] * inv; sc[tt] = a;
    out_id[(long)(t + 1) * BB * LL + b * LL + tt] = (mask[b * LL + tt] == 0) ? 0.f : a;
  }
  __syncthreads();
  float w = 0.f;                                      // weighted[b][tt]
  const _Float16* rbase = route16 + ((long)b * LL) * HH + tt;
  for (int l = 0; l < LL; ++l) w += sc[l] * (float)rbase[(long)l * HH];
  rate_in[(long)b * 512 + tt]      = (_Float16)hb[tt];
  rate_in[(long)b * 512 + HH + tt] = (_Float16)w;
}

// ---------------- per-step: fc2 head ----------------
__global__ void fc2_head(const float* __restrict__ fc1out, const float* __restrict__ fc2_w,
                         const float* __restrict__ fc2_b, float* __restrict__ out_rate, int t) {
  int b = blockIdx.x * blockDim.x + threadIdx.x;
  if (b >= BB) return;
  const float* r = fc1out + (long)b * 512;
  float s = fc2_b[0];
  for (int k = 0; k < 512; ++k) s += r[k] * fc2_w[k];
  out_rate[(long)(t + 1) * BB + b] = 1.f / (1.f + __expf(-s));
}

// ---------------- workspace layout (bytes, all 256-aligned); total ~106 MB ----
static const size_t OFF_NODES16 = 0;                          // 32768*256 f16
static const size_t OFF_XW      = OFF_NODES16 + 16777216;     // 32768*256 f32
static const size_t OFF_AS      = OFF_XW      + 33554432;     // 32768*4 f32
static const size_t OFF_AD      = OFF_AS      + 524288;
static const size_t OFF_ROUTE32 = OFF_AD      + 524288;       // 32768*256 f32
static const size_t OFF_ROUTE16 = OFF_ROUTE32 + 33554432;     // 32768*256 f16
static const size_t OFF_H       = OFF_ROUTE16 + 16777216;     // 256*256 f32
static const size_t OFF_H16     = OFF_H       + 262144;       // 256*256 f16
static const size_t OFF_X16     = OFF_H16     + 131072;       // 256*288 f16
static const size_t OFF_GI      = OFF_X16     + 147456;       // 256*768 f32
static const size_t OFF_GH      = OFF_GI      + 786432;
static const size_t OFF_RATEIN  = OFF_GH      + 786432;       // 256*512 f16
static const size_t OFF_FC1OUT  = OFF_RATEIN  + 262144;       // 256*512 f32
static const size_t OFF_GATWP   = OFF_FC1OUT  + 524288;       // packed 256x256
static const size_t OFF_WIHP    = OFF_GATWP   + 131072;       // packed 288x768
static const size_t OFF_WHHP    = OFF_WIHP    + 442368;       // packed 256x768
static const size_t OFF_FC1P    = OFF_WHHP    + 393216;       // packed 512x512

extern "C" void kernel_launch(void* const* d_in, const int* in_sizes, int n_in,
                              void* d_out, int out_size, void* d_ws, size_t ws_size,
                              hipStream_t stream) {
  const float* route_emb = (const float*)d_in[0];
  const int*   trg_id    = (const int*)  d_in[2];
  const float* trg_rate  = (const float*)d_in[3];
  const int*   mask      = (const int*)  d_in[5];
  const float* emb_id    = (const float*)d_in[6];
  const float* gat_w     = (const float*)d_in[7];
  const float* a_src     = (const float*)d_in[8];
  const float* a_dst     = (const float*)d_in[9];
  const float* gat_bias  = (const float*)d_in[10];
  const float* ln_g      = (const float*)d_in[11];
  const float* ln_b      = (const float*)d_in[12];
  const float* w_ih      = (const float*)d_in[13];
  const float* w_hh      = (const float*)d_in[14];
  const float* b_ih      = (const float*)d_in[15];
  const float* b_hh      = (const float*)d_in[16];
  const float* fc1_w     = (const float*)d_in[17];
  const float* fc1_b     = (const float*)d_in[18];
  const float* fc2_w     = (const float*)d_in[19];
  const float* fc2_b     = (const float*)d_in[20];

  char* ws = (char*)d_ws;
  _Float16* nodes16 = (_Float16*)(ws + OFF_NODES16);
  float*    xw      = (float*)   (ws + OFF_XW);
  float*    as_buf  = (float*)   (ws + OFF_AS);
  float*    ad_buf  = (float*)   (ws + OFF_AD);
  float*    route32 = (float*)   (ws + OFF_ROUTE32);
  _Float16* route16 = (_Float16*)(ws + OFF_ROUTE16);
  float*    hbuf    = (float*)   (ws + OFF_H);
  _Float16* h16     = (_Float16*)(ws + OFF_H16);
  _Float16* x16     = (_Float16*)(ws + OFF_X16);
  float*    gi      = (float*)   (ws + OFF_GI);
  float*    gh      = (float*)   (ws + OFF_GH);
  _Float16* rate_in = (_Float16*)(ws + OFF_RATEIN);
  float*    fc1out  = (float*)   (ws + OFF_FC1OUT);
  _Float16* gatw_p  = (_Float16*)(ws + OFF_GATWP);
  _Float16* wih_p   = (_Float16*)(ws + OFF_WIHP);
  _Float16* whh_p   = (_Float16*)(ws + OFF_WHHP);
  _Float16* fc1_p   = (_Float16*)(ws + OFF_FC1P);

  float* out_id   = (float*)d_out;                       // (T,B,L)
  float* out_rate = (float*)d_out + (long)TT * BB * LL;  // (T,B,1)

  auto gemm = [&](const _Float16* A, int lda, const _Float16* Bp, const float* bias,
                  float* C, int ldc, int M, int N, int Kpad, int relu) {
    int waves = (M / 16) * (N / 64);   // one wave per 16x64 strip
    int blocks = (waves + 3) / 4;
    wmma_gemm4<<<blocks, 128, 0, stream>>>(A, lda, Bp, bias, C, ldc, M, N, Kpad, relu);
  };

  // --- init: zero outputs (slice t=0 must be zero) ---
  long nout = (long)out_size;
  zero_kernel<<<(unsigned)((nout + 255) / 256), 256, 0, stream>>>((float*)d_out, nout);

  // --- one-time conversions / weight packing ---
  long nnodes = (long)NNODE * HH;
  cvt_f32_to_f16<<<(unsigned)((nnodes + 255) / 256), 256, 0, stream>>>(route_emb, nodes16, nnodes);
  pack_weight_f16<<<512, 256, 0, stream>>>(gat_w, gatw_p, 256, 256, 256, 256, 0);
  pack_weight_f16<<<1024, 256, 0, stream>>>(w_ih, wih_p, 257, 288, 768, 257, 1);
  pack_weight_f16<<<1024, 256, 0, stream>>>(w_hh, whh_p, 256, 256, 768, 256, 1);
  pack_weight_f16<<<1024, 256, 0, stream>>>(fc1_w, fc1_p, 512, 512, 512, 512, 1);

  // --- GAT ---
  gemm(nodes16, 256, gatw_p, nullptr, xw, 256, NNODE, 256, 256, 0);
  gat_attn_coef<<<(NNODE * 4 + 255) / 256, 256, 0, stream>>>(xw, a_src, a_dst, as_buf, ad_buf);
  gat_combine_ln<<<NNODE, 256, 0, stream>>>(xw, as_buf, ad_buf, route_emb, gat_bias,
                                            ln_g, ln_b, route32, route16);
  hidden_init<<<(BB * HH + 255) / 256, 256, 0, stream>>>(route32, hbuf, h16);

  // --- sequential GRU scan (255 steps) ---
  for (int t = 0; t < TT - 1; ++t) {
    build_x<<<(BB * 288 + 255) / 256, 256, 0, stream>>>(trg_id, trg_rate, emb_id, x16, t);
    gemm(x16, 288, wih_p, b_ih, gi, 768, BB, 768, 288, 0);
    gemm(h16, 256, whh_p, b_hh, gh, 768, BB, 768, 256, 0);
    gru_combine<<<(BB * HH + 255) / 256, 256, 0, stream>>>(gi, gh, hbuf, h16);
    attn_step<<<BB, 256, 0, stream>>>(hbuf, route16, mask, out_id, rate_in, t);
    gemm(rate_in, 512, fc1_p, fc1_b, fc1out, 512, BB, 512, 512, 1);
    fc2_head<<<1, 256, 0, stream>>>(fc1out, fc2_w, fc2_b, out_rate, t);
  }
}
